// QuantLinear_19911468384456
// MI455X (gfx1250) — matrix operand order repaired and checked by reading, not compiled, not run on hardware
//
#include <hip/hip_runtime.h>

// Problem constants (match reference).
#define BDIM  4
#define SEQ   2048
#define DIN   4096
#define DOUT  4096
#define GSIZE 128
#define MDIM  (BDIM * SEQ)   // 8192

// GEMM tiling.
#define BM 128
#define BN 128
#define BK 32
#define NK (DIN / BK)        // 128 K-steps (even -> clean 2x unroll)
#define LDA 40               // padded LDS stride (bf16 elems) vs bank conflicts
#define LDB 40

typedef __bf16 bf16_t;
typedef __attribute__((ext_vector_type(16))) __bf16 v16bf;
typedef __attribute__((ext_vector_type(4)))  __bf16 v4bf;
typedef __attribute__((ext_vector_type(8)))  float  v8f;
typedef __attribute__((ext_vector_type(4)))  float  v4f;
typedef __attribute__((ext_vector_type(4)))  int    v4i;

union FragU {
    struct { v4i lo; v4i hi; } i;
    v16bf v;
};

// ---------------------------------------------------------------------------
// Kernel 1: fold dequant + both permutations into a transposed bf16 weight:
//   WT[j, in_reorder[i]] = (intweight[i, out_reorder[j]] - z) * s
// so that  out[m, j] = sum_k x[m, k] * WT[j, k]  (+ bias[j]).
// ---------------------------------------------------------------------------
__global__ __launch_bounds__(256) void dequant_permute_kernel(
    const int*   __restrict__ intweight,   // [DIN, DOUT]
    const float* __restrict__ scales,      // [DIN/GSIZE, DOUT]
    const float* __restrict__ zeros,       // [DIN/GSIZE, DOUT]
    const int*   __restrict__ in_reorder,  // [DIN]
    const int*   __restrict__ out_reorder, // [DOUT]
    bf16_t*      __restrict__ wt)          // [DOUT, DIN] (N-major)
{
    const int j = blockIdx.x * blockDim.x + threadIdx.x;  // output column
    const int i = blockIdx.y;                              // input row
    if (j >= DOUT) return;

    const int oj = out_reorder[j];
    const int g  = i >> 7;  // i / GSIZE
    const float s = scales[(size_t)g * DOUT + oj];
    const float z = zeros [(size_t)g * DOUT + oj];
    const float w = ((float)intweight[(size_t)i * DOUT + oj] - z) * s;
    const int ip = in_reorder[i];
    wt[(size_t)j * DIN + ip] = (bf16_t)w;
}

// ---------------------------------------------------------------------------
// Pipelined bf16 WMMA GEMM helpers.
// ---------------------------------------------------------------------------
struct Stage {
    v4f xa[4];   // 128x32 f32 A-tile slice (4 x float4 per thread)
    v4i wb[2];   // 128x32 bf16 B-tile slice (2 x b128 per thread)
};

__device__ __forceinline__ void global_stage_load(
    Stage& st, const float* __restrict__ X, const bf16_t* __restrict__ WT,
    int bm, int bn, int k0, int tid)
{
#pragma unroll
    for (int ii = 0; ii < 4; ++ii) {
        const int e   = ii * 256 + tid;   // 0..1023
        const int row = e >> 3;
        const int cg  = e & 7;
        st.xa[ii] = *(const v4f*)(X + (size_t)(bm + row) * DIN + k0 + cg * 4);
    }
#pragma unroll
    for (int ii = 0; ii < 2; ++ii) {
        const int e  = ii * 256 + tid;    // 0..511
        const int n  = e >> 2;
        const int kg = e & 3;
        st.wb[ii] = *(const v4i*)(WT + (size_t)(bn + n) * DIN + k0 + kg * 8);
    }
}

__device__ __forceinline__ void lds_stage_store(
    const Stage& st, bf16_t* As, bf16_t* Bs, int tid)
{
#pragma unroll
    for (int ii = 0; ii < 4; ++ii) {
        const int e   = ii * 256 + tid;
        const int row = e >> 3;
        const int cg  = e & 7;
        v4bf xb;
#pragma unroll
        for (int q = 0; q < 4; ++q) xb[q] = (bf16_t)st.xa[ii][q];
        *(v4bf*)(&As[row * LDA + cg * 4]) = xb;
    }
#pragma unroll
    for (int ii = 0; ii < 2; ++ii) {
        const int e  = ii * 256 + tid;
        const int n  = e >> 2;
        const int kg = e & 3;
        *(v4i*)(&Bs[n * LDB + kg * 8]) = st.wb[ii];
    }
}

__device__ __forceinline__ void compute_tile(
    const bf16_t* As, const bf16_t* Bs, v8f acc[4][2],
    int arow_base, int ak0, int bcol_base, int bk0)
{
    v16bf afrag[4], bfrag[2];
#pragma unroll
    for (int mt = 0; mt < 4; ++mt) {
        const bf16_t* p = &As[(arow_base + mt * 16) * LDA + ak0];
        FragU u;
        u.i.lo = *(const v4i*)(p);        // K = ak0 .. ak0+7
        u.i.hi = *(const v4i*)(p + 16);   // K = ak0+16 .. ak0+23
        afrag[mt] = u.v;
    }
#pragma unroll
    for (int nt = 0; nt < 2; ++nt) {
        const bf16_t* p = &Bs[(bcol_base + nt * 16) * LDB + bk0];
        FragU u;
        u.i.lo = *(const v4i*)(p);        // K = bk0 .. bk0+7
        u.i.hi = *(const v4i*)(p + 16);   // K = bk0+16 .. bk0+23
        bfrag[nt] = u.v;
    }
#pragma unroll
    for (int mt = 0; mt < 4; ++mt)
#pragma unroll
        for (int nt = 0; nt < 2; ++nt)
            acc[mt][nt] = __builtin_amdgcn_wmma_f32_16x16x32_bf16(
                false, afrag[mt], false, bfrag[nt],
                (short)0, acc[mt][nt], false, false);
}

// ---------------------------------------------------------------------------
// Kernel 2: out[M,N] = X[M,K] * WT[N,K]^T + bias  — double-buffered pipeline.
//   256 threads = 8 wave32 waves; block tile 128x128x32;
//   each wave: 64x32 patch = 4x2 v_wmma_f32_16x16x32_bf16 accumulators.
// ---------------------------------------------------------------------------
__global__ __launch_bounds__(256) void wmma_gemm_kernel(
    const float*  __restrict__ X,    // [MDIM, DIN] f32
    const bf16_t* __restrict__ WT,   // [DOUT, DIN] bf16
    const float*  __restrict__ bias, // [DOUT]
    float*        __restrict__ out)  // [MDIM, DOUT]
{
    __shared__ bf16_t As[2][BM * LDA];
    __shared__ bf16_t Bs[2][BN * LDB];

    const int tid  = threadIdx.x;
    const int lane = tid & 31;
    const int wave = tid >> 5;      // 0..7
    const int wm   = wave >> 2;     // 0..1  -> 64-row half
    const int wn   = wave & 3;      // 0..3  -> 32-col quarter
    const int bm   = blockIdx.y * BM;
    const int bn   = blockIdx.x * BN;

    v8f acc[4][2] = {};

    const int arow_base = wm * 64 + (lane & 15);
    const int ak0       = (lane >> 4) * 8;     // A frag: K chunk select
    const int bcol_base = wn * 32 + (lane & 15);
    const int bk0       = (lane >> 4) * 8;     // B frag: K chunk select

    Stage st;
    // Prologue: stage tile 0 into buffer 0.
    global_stage_load(st, X, WT, bm, bn, 0, tid);
    lds_stage_store(st, As[0], Bs[0], tid);
    __syncthreads();

    // Main pipeline, unrolled by 2 so LDS buffer indices are constants.
    for (int kt = 0; kt < NK; kt += 2) {
        // --- stage even: compute buf0, fill buf1 with tile kt+1 ---
        global_stage_load(st, X, WT, bm, bn, (kt + 1) * BK, tid);   // kt+1 <= 127
        compute_tile(As[0], Bs[0], acc, arow_base, ak0, bcol_base, bk0);
        lds_stage_store(st, As[1], Bs[1], tid);   // loadcnt wait hides under WMMAs
        __syncthreads();

        // --- stage odd: compute buf1, fill buf0 with tile kt+2 ---
        const bool more = (kt + 2) < NK;
        if (more) global_stage_load(st, X, WT, bm, bn, (kt + 2) * BK, tid);
        compute_tile(As[1], Bs[1], acc, arow_base, ak0, bcol_base, bk0);
        if (more) lds_stage_store(st, As[0], Bs[0], tid);
        __syncthreads();
    }

    // ---- epilogue: C/D layout: VGPR v -> M = v (+8 for lanes 16-31), N = lane&15
    const int orow0 = bm + wm * 64;
    const int ocol0 = bn + wn * 32;
#pragma unroll
    for (int mt = 0; mt < 4; ++mt) {
#pragma unroll
        for (int nt = 0; nt < 2; ++nt) {
            const int col  = ocol0 + nt * 16 + (lane & 15);
            const float bv = bias[col];
            const int rbase = orow0 + mt * 16 + ((lane >> 4) << 3);
#pragma unroll
            for (int v = 0; v < 8; ++v) {
                out[(size_t)(rbase + v) * DOUT + col] = acc[mt][nt][v] + bv;
            }
        }
    }
}

// ---------------------------------------------------------------------------
extern "C" void kernel_launch(void* const* d_in, const int* in_sizes, int n_in,
                              void* d_out, int out_size, void* d_ws, size_t ws_size,
                              hipStream_t stream) {
    const float* x           = (const float*)d_in[0];
    const int*   in_reorder  = (const int*)  d_in[1];
    const int*   out_reorder = (const int*)  d_in[2];
    const int*   intweight   = (const int*)  d_in[3];
    const float* scales      = (const float*)d_in[4];
    const float* zeros       = (const float*)d_in[5];
    const float* bias        = (const float*)d_in[6];
    float*       out         = (float*)d_out;

    // Workspace: transposed dequantized weight, DOUT*DIN bf16 = 32 MiB.
    bf16_t* wt = (bf16_t*)d_ws;

    dequant_permute_kernel<<<dim3(DOUT / 256, DIN), 256, 0, stream>>>(
        intweight, scales, zeros, in_reorder, out_reorder, wt);

    wmma_gemm_kernel<<<dim3(DOUT / BN, MDIM / BM), 256, 0, stream>>>(
        x, wt, bias, out);
}
